// VectorQuantizer_2164663517487
// MI455X (gfx1250) — matrix operand order repaired and checked by reading, not compiled, not run on hardware
//
#include <hip/hip_runtime.h>
#include <hip/hip_bf16.h>

typedef __attribute__((ext_vector_type(16))) _Float16 v16h;
typedef __attribute__((ext_vector_type(8)))  _Float16 v8h;
typedef __attribute__((ext_vector_type(8)))  float    v8f;
typedef __attribute__((ext_vector_type(4)))  int      v4i;

// Problem constants from the reference: B=64, C=256, H=W=32, K=1024
#define NPIX      65536      // B*H*W
#define NCH       256
#define NCODE     1024
#define HWSZ      1024       // H*W
#define TILE_CODES 16
#define NTILES    64         // NCODE / TILE_CODES
#define PITCH     264        // 256 + 8 halves padding -> kills LDS bank conflicts
#define PIX_PER_WAVE  32     // two 16-row A tiles per wave
#define PIX_PER_BLOCK 256    // 8 waves * 32 pixels

#if defined(__has_builtin)
#if __has_builtin(__builtin_amdgcn_global_load_async_to_lds_b128)
#define HAVE_ASYNC_LDS 1
#endif
#endif

#ifdef HAVE_ASYNC_LDS
typedef __attribute__((address_space(1))) v4i* gp_v4i;   // global int4*
typedef __attribute__((address_space(3))) v4i* lp_v4i;   // LDS int4*
#endif

// ---------------------------------------------------------------------------
// Kernel 1: convert emb [K,C] f32 -> f16, and compute ee[k] = sum_c emb^2
// ---------------------------------------------------------------------------
__global__ __launch_bounds__(256) void vq_prep_kernel(const float* __restrict__ emb,
                                                      _Float16* __restrict__ embh,
                                                      float* __restrict__ ee) {
    __shared__ float red[256];
    const int k = blockIdx.x;
    const int c = threadIdx.x;
    const float v = emb[(size_t)k * NCH + c];
    embh[(size_t)k * NCH + c] = (_Float16)v;
    red[c] = v * v;
    __syncthreads();
    for (int s = 128; s > 0; s >>= 1) {
        if (c < s) red[c] += red[c + s];
        __syncthreads();
    }
    if (c == 0) ee[k] = red[0];
}

// ---------------------------------------------------------------------------
// Kernel 2: per-pixel argmin over all codes via WMMA f16 (wave32).
// Each wave: 32 pixels (2 A tiles) x all 1024 codes, C=256 contraction.
// B fragments from double-buffered LDS; each B fragment feeds 2 WMMAs.
// ---------------------------------------------------------------------------
__global__ __launch_bounds__(256) void vq_argmin_kernel(const float* __restrict__ x,
                                                        const _Float16* __restrict__ embh,
                                                        const float* __restrict__ ee,
                                                        int* __restrict__ idxOut) {
    __shared__ _Float16 smem[2][TILE_CODES * PITCH];

    const int tid  = threadIdx.x;
    const int wave = tid >> 5;
    const int lane = tid & 31;
    const int grp  = lane >> 4;       // half-wave: 0 or 1
    const int nl   = lane & 15;       // row m for A, column n for B/C/D
    const int pixBase = blockIdx.x * PIX_PER_BLOCK + wave * PIX_PER_WAVE;

    // ---- Load A fragments: 2 tiles x 16 pixels x 256 channels, f32->f16.
    // A k order used here (contiguous, same as B): grp=0 holds K 0..15, grp=1 holds K 16..31.
    v16h afrag[2][8];
    #pragma unroll
    for (int t = 0; t < 2; ++t) {
        const int p  = pixBase + t * 16 + nl;   // this lane's pixel row
        const int b  = p >> 10;                 // / (H*W)
        const int hw = p & (HWSZ - 1);
        const size_t base = (size_t)b * NCH * HWSZ + hw;
        #pragma unroll
        for (int cc = 0; cc < 8; ++cc) {
            const int c0 = cc * 32;
            #pragma unroll
            for (int j = 0; j < 16; ++j) {
                const int k = j + grp * 16;
                afrag[t][cc][j] = (_Float16)x[base + (size_t)(c0 + k) * HWSZ];
            }
        }
    }

    float bestV[2][8];
    int   bestI[2][8];
    #pragma unroll
    for (int t = 0; t < 2; ++t)
        #pragma unroll
        for (int r = 0; r < 8; ++r) { bestV[t][r] = 3.4e38f; bestI[t][r] = 0; }

    // Staging helper: 256 threads move 16 codes x 256 ch (f16, 8 KB) into a buffer.
    const int srow  = tid >> 4;      // code row 0..15
    const int spart = tid & 15;      // 16-half chunk 0..15
    auto stage = [&](int tile, int buf) {
        const _Float16* src = embh + ((size_t)(tile * TILE_CODES + srow) * NCH + spart * 16);
        _Float16* dst = &smem[buf][srow * PITCH + spart * 16];
#ifdef HAVE_ASYNC_LDS
        __builtin_amdgcn_global_load_async_to_lds_b128(
            (gp_v4i)(const void*)src,       (lp_v4i)(void*)dst,       0, 0);
        __builtin_amdgcn_global_load_async_to_lds_b128(
            (gp_v4i)(const void*)(src + 8), (lp_v4i)(void*)(dst + 8), 0, 0);
#else
        ((uint4*)dst)[0] = ((const uint4*)src)[0];
        ((uint4*)dst)[1] = ((const uint4*)src)[1];
#endif
    };
    auto stage_wait = [&]() {
#ifdef HAVE_ASYNC_LDS
        asm volatile("s_wait_asynccnt 0" ::: "memory");
#endif
    };

    stage(0, 0);
    stage_wait();
    __syncthreads();

    for (int tile = 0; tile < NTILES; ++tile) {
        const int cur = tile & 1;
        if (tile + 1 < NTILES) stage(tile + 1, cur ^ 1);   // overlap with compute

        // ---- 2x (16x16) cross tiles: 16 x v_wmma_f32_16x16x32_f16 over C=256
        v8f acc0 = {}, acc1 = {};
        #pragma unroll
        for (int cc = 0; cc < 8; ++cc) {
            const int c0 = cc * 32;
            // B k order used here (contiguous, same as A): grp=0 holds K 0..15, grp=1 holds K 16..31 (col = nl)
            const _Float16* bp = &smem[cur][nl * PITCH + c0 + grp * 16];
            v8h lo = *(const v8h*)bp;
            v8h hi = *(const v8h*)(bp + 8);
            v16h bfrag = __builtin_shufflevector(lo, hi,
                0, 1, 2, 3, 4, 5, 6, 7, 8, 9, 10, 11, 12, 13, 14, 15);
            acc0 = __builtin_amdgcn_wmma_f32_16x16x32_f16(
                false, afrag[0][cc], false, bfrag, (short)0, acc0, false, false);
            acc1 = __builtin_amdgcn_wmma_f32_16x16x32_f16(
                false, afrag[1][cc], false, bfrag, (short)0, acc1, false, false);
        }

        // score = ee[n] - 2*cross   (x_sq constant per row -> dropped).
        // cand strictly increases across tiles for a fixed lane, so a plain
        // strict '<' keeps argmin-first-tie semantics; branchless cndmask selects.
        const float eeV = ee[tile * TILE_CODES + nl];
        const int cand  = tile * TILE_CODES + nl;
        #pragma unroll
        for (int r = 0; r < 8; ++r) {
            const float s0 = fmaf(-2.0f, acc0[r], eeV);
            const bool t0 = s0 < bestV[0][r];
            bestV[0][r] = t0 ? s0  : bestV[0][r];
            bestI[0][r] = t0 ? cand : bestI[0][r];
            const float s1 = fmaf(-2.0f, acc1[r], eeV);
            const bool t1 = s1 < bestV[1][r];
            bestV[1][r] = t1 ? s1  : bestV[1][r];
            bestI[1][r] = t1 ? cand : bestI[1][r];
        }

        stage_wait();      // next buffer fully landed in LDS
        __syncthreads();   // ...and everyone done reading the current one
    }

    // ---- Cross-lane argmin inside each 16-lane group (lexicographic: first-min ties)
    #pragma unroll
    for (int t = 0; t < 2; ++t) {
        #pragma unroll
        for (int r = 0; r < 8; ++r) {
            float v = bestV[t][r];
            int   i = bestI[t][r];
            #pragma unroll
            for (int msk = 1; msk < 16; msk <<= 1) {
                const float v2 = __shfl_xor(v, msk, 32);
                const int   i2 = __shfl_xor(i, msk, 32);
                const bool take = (v2 < v) | ((v2 == v) & (i2 < i));
                v = take ? v2 : v;
                i = take ? i2 : i;
            }
            // C/D layout: VGPR r of lane L -> M = r + 8*(L>>4)
            if (nl == 0) idxOut[pixBase + t * 16 + r + grp * 8] = i;
        }
    }
}

// ---------------------------------------------------------------------------
// Kernel 3: gather out[b,c,h,w] = emb[idx[b,h,w], c]  (float4 coalesced write)
// ---------------------------------------------------------------------------
__global__ __launch_bounds__(256) void vq_gather_kernel(const float* __restrict__ emb,
                                                        const int* __restrict__ idxIn,
                                                        float* __restrict__ out) {
    const size_t gid4 = ((size_t)blockIdx.x * 256 + threadIdx.x) * 4;
    const int hw = (int)(gid4 & (HWSZ - 1));          // multiple of 4
    const int c  = (int)((gid4 >> 10) & (NCH - 1));   // same c for all 4
    const int b  = (int)(gid4 >> 18);
    const int4 kk = *(const int4*)&idxIn[b * HWSZ + hw];
    float4 o;
    o.x = emb[(size_t)kk.x * NCH + c];
    o.y = emb[(size_t)kk.y * NCH + c];
    o.z = emb[(size_t)kk.z * NCH + c];
    o.w = emb[(size_t)kk.w * NCH + c];
    *(float4*)&out[gid4] = o;
}

// ---------------------------------------------------------------------------
extern "C" void kernel_launch(void* const* d_in, const int* in_sizes, int n_in,
                              void* d_out, int out_size, void* d_ws, size_t ws_size,
                              hipStream_t stream) {
    const float* x   = (const float*)d_in[0];   // [64,256,32,32] f32
    const float* emb = (const float*)d_in[1];   // [1024,256]     f32
    float* out = (float*)d_out;                 // [64,256,32,32] f32

    char* ws = (char*)d_ws;
    _Float16* embh = (_Float16*)ws;                         // 512 KB
    float*    ee   = (float*)(ws + 512 * 1024);             //   4 KB
    int*      idx  = (int*)(ws + 512 * 1024 + 4096);        // 256 KB

    vq_prep_kernel<<<dim3(NCODE), dim3(256), 0, stream>>>(emb, embh, ee);
    vq_argmin_kernel<<<dim3(NPIX / PIX_PER_BLOCK), dim3(256), 0, stream>>>(x, embh, ee, idx);
    vq_gather_kernel<<<dim3((NPIX * NCH) / (256 * 4)), dim3(256), 0, stream>>>(emb, idx, out);
    (void)in_sizes; (void)n_in; (void)out_size; (void)ws_size;
}
